// GlobalSSMEncoder_61521111548375
// MI455X (gfx1250) — matrix operand order repaired and checked by reading, not compiled, not run on hardware
//
#include <hip/hip_runtime.h>
#include <hip/hip_bf16.h>

// ---------------- problem constants ----------------
#define B_SZ    4
#define T_LEN   4096
#define D_MODEL 256
#define D_INNER 512
#define D_STATE 16
#define DT_RANK 16
#define D_CONV  4
#define BT      (B_SZ * T_LEN)          // 16384 tokens
#define XZ_DIM  (2 * D_INNER)           // 1024
#define XDBL_N  (DT_RANK + 2 * D_STATE) // 48

// ---------------- vector types for WMMA ----------------
typedef __bf16          bf16x16 __attribute__((ext_vector_type(16)));
typedef unsigned short  u16x8   __attribute__((ext_vector_type(8)));
typedef float           f32x8   __attribute__((ext_vector_type(8)));

union BFVec { bf16x16 b; u16x8 h[2]; };

__device__ __forceinline__ unsigned short f2bf(float f) {
    unsigned u = __float_as_uint(f);
    u += 0x7FFFu + ((u >> 16) & 1u);   // round-to-nearest-even
    return (unsigned short)(u >> 16);
}
__device__ __forceinline__ float sigm(float x) { return 1.0f / (1.0f + __expf(-x)); }

// ---------------- importance gating ----------------
__global__ void imp_max_kernel(const float* __restrict__ imp, float* __restrict__ gmax) {
    __shared__ float red[256];
    int b = blockIdx.x;
    float m = -3.4e38f;
    for (int t = threadIdx.x; t < T_LEN; t += 256)
        m = fmaxf(m, imp[(size_t)b * T_LEN + t]);
    red[threadIdx.x] = m; __syncthreads();
    for (int st = 128; st > 0; st >>= 1) {
        if (threadIdx.x < st) red[threadIdx.x] = fmaxf(red[threadIdx.x], red[threadIdx.x + st]);
        __syncthreads();
    }
    if (threadIdx.x == 0) gmax[b] = red[0];
}

__global__ void gate_kernel(const float* __restrict__ x, const float* __restrict__ imp,
                            const float* __restrict__ gmax, float* __restrict__ h) {
    size_t idx = (size_t)blockIdx.x * blockDim.x + threadIdx.x;   // BT*D_MODEL
    size_t bt  = idx >> 8;      // /256
    size_t b   = bt >> 12;      // /4096
    h[idx] = x[idx] * (imp[bt] / (gmax[b] + 1e-6f));
}

// ---------------- f32 -> bf16 staging ----------------
__global__ void cvt_bf16_kernel(const float* __restrict__ in, unsigned short* __restrict__ out, int n) {
    int i = blockIdx.x * blockDim.x + threadIdx.x;
    if (i < n) out[i] = f2bf(in[i]);
}

// ---------------- WMMA GEMM: C[M,N] = A[M,K] * W[N,K]^T (both K-contiguous, bf16 in, f32 out)
// Register-blocked: one wave32 owns a 16 x (16*NT) strip of C — one A fragment is
// reused across NT B fragments / accumulators, so each K-step issues NT WMMAs for
// 2+2*NT b128 loads (NT=4: 10 loads -> 4 v_wmma, 4x A reuse).
template <int NT>
__global__ void wmma_gemm_bf16(const unsigned short* __restrict__ A,
                               const unsigned short* __restrict__ W,
                               float* __restrict__ C,
                               int M, int N, int K) {
    const int lane = threadIdx.x & 31;
    const int wave = (int)((blockIdx.x * blockDim.x + threadIdx.x) >> 5);
    const int ng   = N / (16 * NT);          // N-strip groups
    const int tm   = wave / ng;
    const int tn0  = (wave - tm * ng) * NT;  // first N tile of this wave's strip
    if (tm >= (M >> 4)) return;              // wave-uniform: EXEC stays all-ones for WMMA

    const int mn = lane & 15;                // row-within-tile (M for A, N for W)
    const int hi = lane >> 4;                // lane-half selector

    const unsigned short* arow = A + (size_t)(tm * 16 + mn) * K;
    const unsigned short* wrow[NT];
    #pragma unroll
    for (int j = 0; j < NT; ++j)
        wrow[j] = W + (size_t)((tn0 + j) * 16 + mn) * K;

    f32x8 acc[NT];
    #pragma unroll
    for (int j = 0; j < NT; ++j) acc[j] = (f32x8){};

    for (int k0 = 0; k0 < K; k0 += 32) {
        BFVec av, bv[NT];
        // A 16x32 bf16 layout: lane-half hi holds K = {k0+8*hi .. +7} and {k0+16+8*hi .. +7}
        av.h[0] = *(const u16x8*)(arow + k0 + hi * 8);
        av.h[1] = *(const u16x8*)(arow + k0 + 16 + hi * 8);
        // B 32x16 bf16 layout: lane-half hi holds K = {k0+8*hi .. +7} and {k0+16+8*hi .. +7}
        #pragma unroll
        for (int j = 0; j < NT; ++j) {
            bv[j].h[0] = *(const u16x8*)(wrow[j] + k0 + hi * 8);
            bv[j].h[1] = *(const u16x8*)(wrow[j] + k0 + 16 + hi * 8);
        }
        if (k0 + 32 < K) __builtin_prefetch(arow + k0 + 32, 0, 0);
        #pragma unroll
        for (int j = 0; j < NT; ++j)
            acc[j] = __builtin_amdgcn_wmma_f32_16x16x32_bf16(false, av.b, false, bv[j].b,
                                                             (short)0, acc[j], false, false);
    }
    // D layout: vgpr r -> M = r + 8*hi, N = lane&15
    #pragma unroll
    for (int j = 0; j < NT; ++j) {
        float* crow = C + (size_t)(tm * 16 + hi * 8) * N + (tn0 + j) * 16 + mn;
        #pragma unroll
        for (int r = 0; r < 8; ++r) crow[(size_t)r * N] = acc[j][r];
    }
}

// ---------------- causal depthwise conv (K=4) + bias + SiLU; emits f32 u and bf16 u ----------------
__global__ void conv_silu_kernel(const float* __restrict__ xz, const float* __restrict__ Wc,
                                 const float* __restrict__ bc, float* __restrict__ u,
                                 unsigned short* __restrict__ u_bf) {
    size_t idx = (size_t)blockIdx.x * blockDim.x + threadIdx.x;  // BT*D_INNER
    int    d   = (int)(idx & (D_INNER - 1));
    size_t bt  = idx >> 9;
    int    t   = (int)(bt & (T_LEN - 1));
    size_t b   = bt >> 12;
    float acc = bc[d];
    #pragma unroll
    for (int k = 0; k < D_CONV; ++k) {
        int tt = t + k - (D_CONV - 1);
        if (tt >= 0)
            acc += xz[((b * T_LEN + tt) * XZ_DIM) + d] * Wc[d * D_CONV + k];
    }
    float s = acc * sigm(acc);   // SiLU
    u[idx]    = s;
    u_bf[idx] = f2bf(s);
}

// ---------------- delta = softplus(dt @ Wdt^T + bdt), K = DT_RANK = 16 ----------------
__global__ void dt_delta_kernel(const float* __restrict__ xdbl, const float* __restrict__ Wdt,
                                const float* __restrict__ bdt, float* __restrict__ dlt) {
    size_t idx = (size_t)blockIdx.x * blockDim.x + threadIdx.x;  // BT*D_INNER
    int    d   = (int)(idx & (D_INNER - 1));
    size_t bt  = idx >> 9;
    const float* dv = xdbl + bt * XDBL_N;          // dt = xdbl[:, :16]
    const float* wr = Wdt + (size_t)d * DT_RANK;
    float acc = bdt[d];
    #pragma unroll
    for (int r = 0; r < DT_RANK; ++r) acc += dv[r] * wr[r];
    dlt[idx] = (acc > 20.0f) ? acc : log1pf(__expf(acc));
}

// ---------------- selective scan: lane = (b,d,s); 16-lane shuffle reduction over s ----------------
// Fuses +u*D_skip and the SiLU(z) gate; emits bf16 y for the out_proj WMMA.
__global__ void scan_kernel(const float* __restrict__ dlt,  const float* __restrict__ u,
                            const float* __restrict__ xdbl, const float* __restrict__ xz,
                            const float* __restrict__ A_log, const float* __restrict__ Dsk,
                            unsigned short* __restrict__ y_bf) {
    int tid = blockIdx.x * blockDim.x + threadIdx.x;
    int s   = tid & 15;
    int g   = tid >> 4;
    if (g >= B_SZ * D_INNER) return;
    int d = g & (D_INNER - 1);
    int b = g >> 9;

    const float Ac = -__expf(A_log[d * D_STATE + s]);   // A = -exp(A_log)
    const float Dv = Dsk[d];
    float h = 0.0f;
    const size_t bt0 = (size_t)b * T_LEN;

    for (int t = 0; t < T_LEN; ++t) {
        size_t bt = bt0 + t;
        float de = dlt[bt * D_INNER + d];
        float uu = u[bt * D_INNER + d];
        float Bv = xdbl[bt * XDBL_N + DT_RANK + s];
        float Cv = xdbl[bt * XDBL_N + DT_RANK + D_STATE + s];
        h = __expf(de * Ac) * h + (de * uu) * Bv;
        float yv = h * Cv;
        yv += __shfl_xor(yv, 1, 16);
        yv += __shfl_xor(yv, 2, 16);
        yv += __shfl_xor(yv, 4, 16);
        yv += __shfl_xor(yv, 8, 16);
        if (s == 0) {
            float zv = xz[bt * XZ_DIM + D_INNER + d];
            float o  = (yv + uu * Dv) * (zv * sigm(zv));
            y_bf[bt * D_INNER + d] = f2bf(o);
        }
    }
}

// ---------------- final LayerNorm over D_MODEL ----------------
__global__ void ln_kernel(const float* __restrict__ h, const float* __restrict__ g,
                          const float* __restrict__ beta, float* __restrict__ out) {
    __shared__ float red[256];
    int row = blockIdx.x, tid = threadIdx.x;
    float v = h[(size_t)row * D_MODEL + tid];
    red[tid] = v; __syncthreads();
    for (int st = 128; st > 0; st >>= 1) { if (tid < st) red[tid] += red[tid + st]; __syncthreads(); }
    float mu = red[0] * (1.0f / D_MODEL); __syncthreads();
    float dv = v - mu;
    red[tid] = dv * dv; __syncthreads();
    for (int st = 128; st > 0; st >>= 1) { if (tid < st) red[tid] += red[tid + st]; __syncthreads(); }
    float var = red[0] * (1.0f / D_MODEL);
    out[(size_t)row * D_MODEL + tid] = dv * rsqrtf(var + 1e-5f) * g[tid] + beta[tid];
}

// ---------------- host launcher ----------------
extern "C" void kernel_launch(void* const* d_in, const int* in_sizes, int n_in,
                              void* d_out, int out_size, void* d_ws, size_t ws_size,
                              hipStream_t stream) {
    (void)in_sizes; (void)n_in; (void)out_size; (void)ws_size;
    const float* x_in    = (const float*)d_in[0];
    const float* imp     = (const float*)d_in[1];
    const float* Wi_all  = (const float*)d_in[2];   // (3, 1024, 256)
    const float* Wc_all  = (const float*)d_in[3];   // (3, 512, 4)
    const float* bc_all  = (const float*)d_in[4];   // (3, 512)
    const float* Wx_all  = (const float*)d_in[5];   // (3, 48, 512)
    const float* Wdt_all = (const float*)d_in[6];   // (3, 512, 16)
    const float* bdt_all = (const float*)d_in[7];   // (3, 512)
    const float* Al_all  = (const float*)d_in[8];   // (3, 512, 16)
    const float* Dk_all  = (const float*)d_in[9];   // (3, 512)
    const float* Wo_all  = (const float*)d_in[10];  // (3, 256, 512)
    const float* lng     = (const float*)d_in[11];
    const float* lnb     = (const float*)d_in[12];
    float* out = (float*)d_out;

    // workspace carve-out
    size_t off = 0;
    char* base = (char*)d_ws;
    auto carve = [&](size_t bytes) -> char* {
        char* p = base + off;
        off += (bytes + 255) & ~(size_t)255;
        return p;
    };
    float* hA   = (float*)carve((size_t)BT * D_MODEL * 4);
    float* hB   = (float*)carve((size_t)BT * D_MODEL * 4);
    float* xz   = (float*)carve((size_t)BT * XZ_DIM * 4);
    float* ub   = (float*)carve((size_t)BT * D_INNER * 4);
    float* xdbl = (float*)carve((size_t)BT * XDBL_N * 4);
    float* dlt  = (float*)carve((size_t)BT * D_INNER * 4);
    unsigned short* h_bf  = (unsigned short*)carve((size_t)BT * D_MODEL * 2);
    unsigned short* u_bf  = (unsigned short*)carve((size_t)BT * D_INNER * 2);
    unsigned short* y_bf  = (unsigned short*)carve((size_t)BT * D_INNER * 2);
    unsigned short* Wi_bf = (unsigned short*)carve((size_t)XZ_DIM * D_MODEL * 2);
    unsigned short* Wx_bf = (unsigned short*)carve((size_t)XDBL_N * D_INNER * 2);
    unsigned short* Wo_bf = (unsigned short*)carve((size_t)D_MODEL * D_INNER * 2);
    float* gmax = (float*)carve(4 * 4);

    // importance gating: h = x * imp / (max_t imp + 1e-6)
    imp_max_kernel<<<B_SZ, 256, 0, stream>>>(imp, gmax);
    gate_kernel<<<(BT * D_MODEL) / 256, 256, 0, stream>>>(x_in, imp, gmax, hA);

    float* hin = hA;
    float* hout = hB;

    for (int L = 0; L < 3; ++L) {
        const float* Wi  = Wi_all  + (size_t)L * XZ_DIM * D_MODEL;
        const float* Wc  = Wc_all  + (size_t)L * D_INNER * D_CONV;
        const float* bc  = bc_all  + (size_t)L * D_INNER;
        const float* Wx  = Wx_all  + (size_t)L * XDBL_N * D_INNER;
        const float* Wdt = Wdt_all + (size_t)L * D_INNER * DT_RANK;
        const float* bdt = bdt_all + (size_t)L * D_INNER;
        const float* Al  = Al_all  + (size_t)L * D_INNER * D_STATE;
        const float* Dk  = Dk_all  + (size_t)L * D_INNER;
        const float* Wo  = Wo_all  + (size_t)L * D_MODEL * D_INNER;

        // stage bf16 operands
        cvt_bf16_kernel<<<(BT * D_MODEL) / 256, 256, 0, stream>>>(hin, h_bf, BT * D_MODEL);
        cvt_bf16_kernel<<<(XZ_DIM * D_MODEL) / 256, 256, 0, stream>>>(Wi, Wi_bf, XZ_DIM * D_MODEL);
        cvt_bf16_kernel<<<(XDBL_N * D_INNER) / 256, 256, 0, stream>>>(Wx, Wx_bf, XDBL_N * D_INNER);
        cvt_bf16_kernel<<<(D_MODEL * D_INNER) / 256, 256, 0, stream>>>(Wo, Wo_bf, D_MODEL * D_INNER);

        // in_proj: xz[BT,1024] = h[BT,256] x Wi[1024,256]^T   (16x64 strips per wave)
        {
            int waves = (BT / 16) * (XZ_DIM / 64);     // 16384 waves
            wmma_gemm_bf16<4><<<waves * 32 / 256, 256, 0, stream>>>(h_bf, Wi_bf, xz, BT, XZ_DIM, D_MODEL);
        }
        // causal conv + SiLU on u half
        conv_silu_kernel<<<(BT * D_INNER) / 256, 256, 0, stream>>>(xz, Wc, bc, ub, u_bf);
        // x_proj: xdbl[BT,48] = u[BT,512] x Wx[48,512]^T      (16x48 strips per wave)
        {
            int waves = (BT / 16) * (XDBL_N / 48);     // 1024 waves
            wmma_gemm_bf16<3><<<waves * 32 / 256, 256, 0, stream>>>(u_bf, Wx_bf, xdbl, BT, XDBL_N, D_INNER);
        }
        // delta = softplus(dt @ Wdt^T + bdt)
        dt_delta_kernel<<<(BT * D_INNER) / 256, 256, 0, stream>>>(xdbl, Wdt, bdt, dlt);
        // selective scan (fused +u*D and SiLU(z) gate), emits bf16 y
        scan_kernel<<<(B_SZ * D_INNER * D_STATE) / 256, 256, 0, stream>>>(
            dlt, ub, xdbl, xz, Al, Dk, y_bf);
        // out_proj: hout[BT,256] = y[BT,512] x Wo[256,512]^T  (16x64 strips per wave)
        {
            int waves = (BT / 16) * (D_MODEL / 64);    // 4096 waves
            wmma_gemm_bf16<4><<<waves * 32 / 256, 256, 0, stream>>>(y_bf, Wo_bf, hout, BT, D_MODEL, D_INNER);
        }
        float* tmp = hin; hin = hout; hout = tmp;
    }

    // final LayerNorm over D_MODEL -> d_out
    ln_kernel<<<BT, 256, 0, stream>>>(hin, lng, lnb, out);
}
